// TemporalAttention_51780125721406
// MI455X (gfx1250) — compile-verified, matrix operand order checked by reading, not run on hardware
//
#include <hip/hip_runtime.h>
#include <hip/hip_bf16.h>

typedef __bf16 bf16_t;
typedef __attribute__((ext_vector_type(8)))  __bf16 v8bf;
typedef __attribute__((ext_vector_type(16))) __bf16 v16bf;
typedef __attribute__((ext_vector_type(8)))  float  v8f;

#define DEV __device__ __forceinline__

constexpr int CB    = 4;
constexpr int CT    = 8, CH = 28, CW = 28;
constexpr int NTOK  = 1 + CT * CH * CW;      // 6273
constexpr int CDIM  = 768;
constexpr int NHEAD = 12;
constexpr int CHD   = 64;
constexpr int TP = 8, HP = 14, WP = 14;
constexpr int NQ    = 1 + TP * HP * WP;      // 1569
constexpr float LNEPS = 1e-5f;

// ---- WMMA helpers ------------------------------------------------------
DEV v8f wmma_bf(v16bf a, v16bf b, v8f c) {
    return __builtin_amdgcn_wmma_f32_16x16x32_bf16(
        false, a, false, b, (short)0, c, false, false);
}

// Load one 16x32 A/B fragment for this lane: two contiguous 8-element
// (16-byte) chunks at p and p+16, per the CDNA5 16-bit A-matrix layout.
DEV v16bf ld_frag(const bf16_t* p) {
    v16bf r;
    v8bf lo = *(const v8bf*)(p);
    v8bf hi = *(const v8bf*)(p + 16);
#pragma unroll
    for (int i = 0; i < 8; i++) { r[i] = lo[i]; r[8 + i] = hi[i]; }
    return r;
}

// ---- Kernel 1: tiled fp32 [K][N] -> bf16 [N][K] transpose --------------
__global__ __launch_bounds__(256)
void transpose_cvt_kernel(const float* __restrict__ src, // [K][N] fp32
                          bf16_t* __restrict__ dst,      // [N][K] bf16
                          int K, int N) {
    __shared__ float tile[32][33];
    const int bx = blockIdx.x * 32;      // N offset
    const int by = blockIdx.y * 32;      // K offset
    const int tx = threadIdx.x & 31;
    const int ty = threadIdx.x >> 5;     // 0..7
#pragma unroll
    for (int j = 0; j < 4; j++)
        tile[ty + j * 8][tx] = src[(size_t)(by + ty + j * 8) * N + bx + tx];
    __syncthreads();
#pragma unroll
    for (int j = 0; j < 4; j++)
        dst[(size_t)(bx + ty + j * 8) * K + by + tx] =
            (bf16_t)tile[tx][ty + j * 8];
}

// ---- Kernel 2: QKV GEMM  [B*N,768] x [768,2304] + bias -----------------
// Block tile 64(M) x 128(N), 8 waves each computing 32x32 (2x2 WMMA tiles).
// Software-pipelined: next k-tile global loads overlap WMMA compute.
__global__ __launch_bounds__(256)
void qkv_gemm_kernel(const float*  __restrict__ X,    // [B*NTOK, 768] fp32
                     const bf16_t* __restrict__ Wt,   // [2304][768] bf16 (transposed)
                     const float*  __restrict__ bias, // [2304]
                     bf16_t*       __restrict__ raw)  // [3][B][NH][NTOK][64]
{
    constexpr int MT = CB * NTOK;    // 25092
    constexpr int KK = CDIM;         // 768

    __shared__ alignas(16) bf16_t lA[64 * 32];
    __shared__ alignas(16) bf16_t lB[128 * 32];

    const int tid  = threadIdx.x;
    const int lane = tid & 31;
    const int wid  = tid >> 5;
    const int wm   = wid & 1;
    const int wn   = wid >> 1;
    const int bm   = blockIdx.x * 64;
    const int bn   = blockIdx.y * 128;
    const int rl   = lane & 15;
    const int kb   = (lane >> 4) * 8;

    const int rA  = tid >> 2;            // A staging: row 0..63
    const int cA  = (tid & 3) * 8;       // A staging: k chunk
    const int nB  = tid & 127;           // B staging: n 0..127
    const int kB0 = (tid >> 7) * 16;     // B staging: k half

    int gr = bm + rA;
    if (gr >= MT) gr = MT - 1;           // clamp; OOB rows never stored
    const float*  aBase = X  + (size_t)gr * KK + cA;
    const bf16_t* bBase = Wt + (size_t)(bn + nB) * KK + kB0;

    float4 a0 = *(const float4*)(aBase);
    float4 a1 = *(const float4*)(aBase + 4);
    v8bf   b0 = *(const v8bf*)(bBase);
    v8bf   b1 = *(const v8bf*)(bBase + 8);

    v8f acc[2][2] = {};

    for (int k0 = 0; k0 < KK; k0 += 32) {
        // commit prefetched registers to LDS
        v8bf apk;
        apk[0] = (bf16_t)a0.x; apk[1] = (bf16_t)a0.y;
        apk[2] = (bf16_t)a0.z; apk[3] = (bf16_t)a0.w;
        apk[4] = (bf16_t)a1.x; apk[5] = (bf16_t)a1.y;
        apk[6] = (bf16_t)a1.z; apk[7] = (bf16_t)a1.w;
        *(v8bf*)(lA + rA * 32 + cA)       = apk;
        *(v8bf*)(lB + nB * 32 + kB0)      = b0;
        *(v8bf*)(lB + nB * 32 + kB0 + 8)  = b1;
        __syncthreads();

        // issue next k-tile global loads (overlap with WMMA below)
        int kn = k0 + 32;
        if (kn < KK) {
            const float* an = aBase + kn;
            __builtin_prefetch(an + 32, 0, 3);
            a0 = *(const float4*)(an);
            a1 = *(const float4*)(an + 4);
            const bf16_t* bp = bBase + kn;
            __builtin_prefetch(bp + 32, 0, 3);
            b0 = *(const v8bf*)(bp);
            b1 = *(const v8bf*)(bp + 8);
        }

        v16bf af[2], bfg[2];
#pragma unroll
        for (int i = 0; i < 2; i++)
            af[i] = ld_frag(lA + (wm * 32 + i * 16 + rl) * 32 + kb);
#pragma unroll
        for (int j = 0; j < 2; j++)
            bfg[j] = ld_frag(lB + (wn * 32 + j * 16 + rl) * 32 + kb);
#pragma unroll
        for (int i = 0; i < 2; i++)
#pragma unroll
            for (int j = 0; j < 2; j++)
                acc[i][j] = wmma_bf(af[i], bfg[j], acc[i][j]);
        __syncthreads();
    }

    // epilogue: bias + scatter into q/k/v raw layout (bf16)
    const int rhalf = (lane >> 4) * 8;
#pragma unroll
    for (int i = 0; i < 2; i++)
#pragma unroll
        for (int j = 0; j < 2; j++) {
            int nl = bn + wn * 32 + j * 16 + rl;
            float bv = bias[nl];
            int which = nl / CDIM;
            int rem   = nl % CDIM;
            int head  = rem / CHD;
            int dch   = rem % CHD;
#pragma unroll
            for (int r = 0; r < 8; r++) {
                int ml = bm + wm * 32 + i * 16 + rhalf + r;
                if (ml < MT) {
                    int b   = ml / NTOK;
                    int tok = ml % NTOK;
                    size_t idx = ((((size_t)which * CB + b) * NHEAD + head)
                                  * NTOK + tok) * CHD + dch;
                    raw[idx] = (bf16_t)(acc[i][j][r] + bv);
                }
            }
        }
}

// ---- Kernel 3: depthwise 3x3x3 pooling conv + LayerNorm ----------------
__global__ __launch_bounds__(64)
void pool_ln_kernel(const bf16_t* __restrict__ raw,
                    const float* __restrict__ wq, const float* __restrict__ wk,
                    const float* __restrict__ wv,
                    const float* __restrict__ gq, const float* __restrict__ bq,
                    const float* __restrict__ gk, const float* __restrict__ bk,
                    const float* __restrict__ gv, const float* __restrict__ bv,
                    bf16_t* __restrict__ pooled)   // [3][B][NH][NQ][64]
{
    const int d     = threadIdx.x;
    const int tok   = blockIdx.x;
    const int bh    = blockIdx.y;
    const int which = blockIdx.z;
    const int b = bh / NHEAD, h = bh % NHEAD;

    const bf16_t* src = raw + (((size_t)which * CB + b) * NHEAD + h)
                              * (size_t)NTOK * CHD;
    const float* wc = which == 0 ? wq : (which == 1 ? wk : wv);
    const float* g  = which == 0 ? gq : (which == 1 ? gk : gv);
    const float* be = which == 0 ? bq : (which == 1 ? bk : bv);

    float val;
    if (tok == 0) {
        val = (float)src[d];
    } else {
        int to = tok - 1;
        int t = to / (HP * WP);
        int r = to % (HP * WP);
        int y = r / WP, x = r % WP;
        float s = 0.f;
#pragma unroll
        for (int kt = 0; kt < 3; kt++) {
            int tin = t + kt - 1;
            if (tin < 0 || tin >= CT) continue;
#pragma unroll
            for (int ky = 0; ky < 3; ky++) {
                int yin = y * 2 + ky - 1;
                if (yin < 0 || yin >= CH) continue;
#pragma unroll
                for (int kx = 0; kx < 3; kx++) {
                    int xin = x * 2 + kx - 1;
                    if (xin < 0 || xin >= CW) continue;
                    int it = 1 + (tin * CH + yin) * CW + xin;
                    s += (float)src[(size_t)it * CHD + d]
                         * wc[d * 27 + (kt * 3 + ky) * 3 + kx];
                }
            }
        }
        val = s;
    }

    __shared__ float sm[64];
    sm[d] = val;
    __syncthreads();
#pragma unroll
    for (int off = 32; off > 0; off >>= 1) {
        if (d < off) sm[d] += sm[d + off];
        __syncthreads();
    }
    float mu = sm[0] * (1.f / 64.f);
    __syncthreads();
    float dv = val - mu;
    sm[d] = dv * dv;
    __syncthreads();
#pragma unroll
    for (int off = 32; off > 0; off >>= 1) {
        if (d < off) sm[d] += sm[d + off];
        __syncthreads();
    }
    float var = sm[0] * (1.f / 64.f);
    float o = dv * rsqrtf(var + LNEPS) * g[d] + be[d];
    pooled[(((size_t)which * CB + b) * NHEAD + h) * (size_t)NQ * CHD
           + (size_t)tok * CHD + d] = (bf16_t)o;
}

// ---- Kernel 4: flash-style attention (WMMA QK^T and PV) ----------------
__global__ __launch_bounds__(128)
void attn_kernel(const bf16_t* __restrict__ pooled,
                 bf16_t* __restrict__ aout)        // [B][NQ][768]
{
    constexpr float SCALE = 0.125f;                // 64^-0.5
    __shared__ alignas(16) bf16_t Kt[32 * 64];     // [tok][d]
    __shared__ alignas(16) bf16_t Vt[64 * 32];     // transposed [d][tok]
    __shared__ alignas(16) bf16_t Pb[4][16 * 32];  // per-wave P tile

    const int tid  = threadIdx.x;
    const int lane = tid & 31;
    const int wid  = tid >> 5;                     // 4 waves
    const int bh   = blockIdx.y;
    const int b = bh / NHEAD, h = bh % NHEAD;
    const int qbase = blockIdx.x * 64;

    const size_t hd = ((size_t)b * NHEAD + h) * (size_t)NQ * CHD;
    const bf16_t* qp = pooled + 0 * (size_t)CB * NHEAD * NQ * CHD + hd;
    const bf16_t* kp = pooled + 1 * (size_t)CB * NHEAD * NQ * CHD + hd;
    const bf16_t* vp = pooled + 2 * (size_t)CB * NHEAD * NQ * CHD + hd;

    const int rl = lane & 15;
    const int kb = (lane >> 4) * 8;

    // Q fragments (16 rows x 64 d, two k-chunks)
    v16bf qa[2];
    {
        int row = qbase + wid * 16 + rl;
        if (row >= NQ) row = NQ - 1;
#pragma unroll
        for (int c = 0; c < 2; c++)
            qa[c] = ld_frag(qp + (size_t)row * CHD + c * 32 + kb);
    }

    v8f o0 = {}, o1 = {}, o2 = {}, o3 = {};
    float mrow[8], lrow[8];
#pragma unroll
    for (int i = 0; i < 8; i++) { mrow[i] = -1e30f; lrow[i] = 0.f; }

    const int sj   = tid >> 2;           // staging token 0..31
    const int sseg = (tid & 3) * 16;     // staging d chunk
    const int NKT  = (NQ + 31) / 32;     // 50

    for (int ktile = 0; ktile < NKT; ktile++) {
        __syncthreads();
        {
            int tok = ktile * 32 + sj;
            int tc  = tok < NQ ? tok : NQ - 1;
            const bf16_t* ks = kp + (size_t)tc * CHD + sseg;
            const bf16_t* vs = vp + (size_t)tc * CHD + sseg;
            // K tile: straight vectorized copy
            v8bf kv0 = *(const v8bf*)(ks);
            v8bf kv1 = *(const v8bf*)(ks + 8);
            *(v8bf*)(Kt + sj * 64 + sseg)     = kv0;
            *(v8bf*)(Kt + sj * 64 + sseg + 8) = kv1;
            // V tile: transpose scatter into [d][tok]
            v8bf vv0 = *(const v8bf*)(vs);
            v8bf vv1 = *(const v8bf*)(vs + 8);
#pragma unroll
            for (int i = 0; i < 8; i++) {
                Vt[(sseg + i) * 32 + sj]     = vv0[i];
                Vt[(sseg + 8 + i) * 32 + sj] = vv1[i];
            }
        }
        __syncthreads();

        // S = Q K^T  (two 16x16 tiles covering 32 k-tokens)
        v8f s0 = {}, s1 = {};
#pragma unroll
        for (int c = 0; c < 2; c++) {
            v16bf bk0 = ld_frag(Kt + (0 * 16 + rl) * 64 + c * 32 + kb);
            v16bf bk1 = ld_frag(Kt + (1 * 16 + rl) * 64 + c * 32 + kb);
            s0 = wmma_bf(qa[c], bk0, s0);
            s1 = wmma_bf(qa[c], bk1, s1);
        }

        bool msk0 = (ktile * 32 + rl) >= NQ;
        bool msk1 = (ktile * 32 + 16 + rl) >= NQ;

        float p0[8], p1[8], mn[8], cexp[8];
#pragma unroll
        for (int i = 0; i < 8; i++) {
            float a  = msk0 ? -1e30f : s0[i] * SCALE;
            float c2 = msk1 ? -1e30f : s1[i] * SCALE;
            p0[i] = a; p1[i] = c2;
            float mx = fmaxf(a, c2);
#pragma unroll
            for (int off = 1; off < 16; off <<= 1)
                mx = fmaxf(mx, __shfl_xor(mx, off, 16));
            mn[i] = fmaxf(mrow[i], mx);
        }
#pragma unroll
        for (int i = 0; i < 8; i++) {
            cexp[i] = __expf(mrow[i] - mn[i]);
            p0[i] = __expf(p0[i] - mn[i]);
            p1[i] = __expf(p1[i] - mn[i]);
            float sum = p0[i] + p1[i];
#pragma unroll
            for (int off = 1; off < 16; off <<= 1)
                sum += __shfl_xor(sum, off, 16);
            lrow[i] = lrow[i] * cexp[i] + sum;
            mrow[i] = mn[i];
            o0[i] *= cexp[i]; o1[i] *= cexp[i];
            o2[i] *= cexp[i]; o3[i] *= cexp[i];
        }

        // repack P (C layout -> A layout) through wave-private LDS
        bf16_t* Pw = Pb[wid];
        {
            int prow = (lane < 16) ? 0 : 8;
#pragma unroll
            for (int i = 0; i < 8; i++) {
                Pw[(prow + i) * 32 + rl]      = (bf16_t)p0[i];
                Pw[(prow + i) * 32 + 16 + rl] = (bf16_t)p1[i];
            }
        }
        asm volatile("s_wait_dscnt 0" ::: "memory");
        v16bf pa = ld_frag(Pw + rl * 32 + kb);

        v16bf bv0 = ld_frag(Vt + (0 * 16 + rl) * 32 + kb);
        v16bf bv1 = ld_frag(Vt + (1 * 16 + rl) * 32 + kb);
        v16bf bv2 = ld_frag(Vt + (2 * 16 + rl) * 32 + kb);
        v16bf bv3 = ld_frag(Vt + (3 * 16 + rl) * 32 + kb);
        o0 = wmma_bf(pa, bv0, o0);
        o1 = wmma_bf(pa, bv1, o1);
        o2 = wmma_bf(pa, bv2, o2);
        o3 = wmma_bf(pa, bv3, o3);
    }

    // epilogue: O / l -> bf16 [b][tok][h*64 + d]
    const int rowl = (lane < 16) ? 0 : 8;
#pragma unroll
    for (int i = 0; i < 8; i++) {
        int tok = qbase + wid * 16 + rowl + i;
        if (tok < NQ) {
            float inv = 1.f / lrow[i];
            size_t base = ((size_t)b * NQ + tok) * CDIM + h * CHD;
            aout[base + 0 * 16 + rl] = (bf16_t)(o0[i] * inv);
            aout[base + 1 * 16 + rl] = (bf16_t)(o1[i] * inv);
            aout[base + 2 * 16 + rl] = (bf16_t)(o2[i] * inv);
            aout[base + 3 * 16 + rl] = (bf16_t)(o3[i] * inv);
        }
    }
}

// ---- Kernel 5: projection GEMM  [B*NQ,768] x [768,768] + bias -> fp32 --
__global__ __launch_bounds__(256)
void proj_gemm_kernel(const bf16_t* __restrict__ A,    // [B*NQ, 768] bf16
                      const bf16_t* __restrict__ Wt,   // [768][768] bf16 (transposed)
                      const float*  __restrict__ bias, // [768]
                      float*        __restrict__ out)  // [B*NQ, 768]
{
    constexpr int MT = CB * NQ;      // 6276
    constexpr int KK = CDIM;
    constexpr int NN = CDIM;

    __shared__ alignas(16) bf16_t lA[64 * 32];
    __shared__ alignas(16) bf16_t lB[128 * 32];

    const int tid  = threadIdx.x;
    const int lane = tid & 31;
    const int wid  = tid >> 5;
    const int wm   = wid & 1;
    const int wn   = wid >> 1;
    const int bm   = blockIdx.x * 64;
    const int bn   = blockIdx.y * 128;
    const int rl   = lane & 15;
    const int kb   = (lane >> 4) * 8;

    const int rA  = tid >> 2;
    const int cA  = (tid & 3) * 8;
    const int nB  = tid & 127;
    const int kB0 = (tid >> 7) * 16;

    int gr = bm + rA;
    if (gr >= MT) gr = MT - 1;
    const bf16_t* aBase = A  + (size_t)gr * KK + cA;
    const bf16_t* bBase = Wt + (size_t)(bn + nB) * KK + kB0;

    v8bf a0 = *(const v8bf*)(aBase);
    v8bf b0 = *(const v8bf*)(bBase);
    v8bf b1 = *(const v8bf*)(bBase + 8);

    v8f acc[2][2] = {};

    for (int k0 = 0; k0 < KK; k0 += 32) {
        *(v8bf*)(lA + rA * 32 + cA)      = a0;
        *(v8bf*)(lB + nB * 32 + kB0)     = b0;
        *(v8bf*)(lB + nB * 32 + kB0 + 8) = b1;
        __syncthreads();

        int kn = k0 + 32;
        if (kn < KK) {
            const bf16_t* an = aBase + kn;
            __builtin_prefetch(an + 32, 0, 3);
            a0 = *(const v8bf*)(an);
            const bf16_t* bp = bBase + kn;
            __builtin_prefetch(bp + 32, 0, 3);
            b0 = *(const v8bf*)(bp);
            b1 = *(const v8bf*)(bp + 8);
        }

        v16bf af[2], bfg[2];
#pragma unroll
        for (int i = 0; i < 2; i++)
            af[i] = ld_frag(lA + (wm * 32 + i * 16 + rl) * 32 + kb);
#pragma unroll
        for (int j = 0; j < 2; j++)
            bfg[j] = ld_frag(lB + (wn * 32 + j * 16 + rl) * 32 + kb);
#pragma unroll
        for (int i = 0; i < 2; i++)
#pragma unroll
            for (int j = 0; j < 2; j++)
                acc[i][j] = wmma_bf(af[i], bfg[j], acc[i][j]);
        __syncthreads();
    }

    const int rhalf = (lane >> 4) * 8;
#pragma unroll
    for (int i = 0; i < 2; i++)
#pragma unroll
        for (int j = 0; j < 2; j++) {
            int nl = bn + wn * 32 + j * 16 + rl;
            float bv = bias[nl];
#pragma unroll
            for (int r = 0; r < 8; r++) {
                int ml = bm + wm * 32 + i * 16 + rhalf + r;
                if (ml < MT)
                    out[(size_t)ml * NN + nl] = acc[i][j][r] + bv;
            }
        }
}

// ---- host launch -------------------------------------------------------
extern "C" void kernel_launch(void* const* d_in, const int* in_sizes, int n_in,
                              void* d_out, int out_size, void* d_ws,
                              size_t ws_size, hipStream_t stream) {
    (void)in_sizes; (void)n_in; (void)out_size; (void)ws_size;
    const float* x      = (const float*)d_in[0];
    const float* qkv_w  = (const float*)d_in[1];
    const float* qkv_b  = (const float*)d_in[2];
    const float* proj_w = (const float*)d_in[3];
    const float* proj_b = (const float*)d_in[4];
    const float* wq     = (const float*)d_in[5];
    const float* wk     = (const float*)d_in[6];
    const float* wv     = (const float*)d_in[7];
    const float* gq     = (const float*)d_in[8];
    const float* bq     = (const float*)d_in[9];
    const float* gk     = (const float*)d_in[10];
    const float* bk     = (const float*)d_in[11];
    const float* gv     = (const float*)d_in[12];
    const float* bv     = (const float*)d_in[13];
    float* out = (float*)d_out;

    char* ws = (char*)d_ws;
    size_t off = 0;
    auto alloc = [&](size_t bytes) {
        size_t o = off;
        off = (off + bytes + 255) & ~(size_t)255;
        return o;
    };
    bf16_t* qkvw_t  = (bf16_t*)(ws + alloc((size_t)CDIM * 3 * CDIM * 2));
    bf16_t* projw_t = (bf16_t*)(ws + alloc((size_t)CDIM * CDIM * 2));
    bf16_t* raw     = (bf16_t*)(ws + alloc((size_t)3 * CB * NHEAD * NTOK * CHD * 2));
    bf16_t* pooled  = (bf16_t*)(ws + alloc((size_t)3 * CB * NHEAD * NQ * CHD * 2));
    bf16_t* aout    = (bf16_t*)(ws + alloc((size_t)CB * NQ * CDIM * 2));

    // one-time weight transpose+convert: [K][N] fp32 -> [N][K] bf16
    dim3 t1((3 * CDIM) / 32, CDIM / 32);
    transpose_cvt_kernel<<<t1, 256, 0, stream>>>(qkv_w, qkvw_t, CDIM, 3 * CDIM);
    dim3 t2(CDIM / 32, CDIM / 32);
    transpose_cvt_kernel<<<t2, 256, 0, stream>>>(proj_w, projw_t, CDIM, CDIM);

    dim3 g1((CB * NTOK + 63) / 64, (3 * CDIM) / 128);
    qkv_gemm_kernel<<<g1, 256, 0, stream>>>(x, qkvw_t, qkv_b, raw);

    dim3 g2(NQ, CB * NHEAD, 3);
    pool_ln_kernel<<<g2, 64, 0, stream>>>(raw, wq, wk, wv,
                                          gq, bq, gk, bk, gv, bv, pooled);

    dim3 g3((NQ + 63) / 64, CB * NHEAD);
    attn_kernel<<<g3, 128, 0, stream>>>(pooled, aout);

    dim3 g4((CB * NQ + 63) / 64, CDIM / 128);
    proj_gemm_kernel<<<g4, 256, 0, stream>>>(aout, projw_t, proj_b, out);
}
